// AttentionBlock_63711544868868
// MI455X (gfx1250) — compile-verified, matrix operand order checked by reading, not run on hardware
//
#include <hip/hip_runtime.h>
#include <hip/hip_bf16.h>
#include <stdint.h>

// ---------------------------------------------------------------------------
// MI455X (gfx1250) attention block: GroupNorm -> QKV 1x1conv -> QK^T softmax
// -> V*P^T -> output 1x1conv + residual.  GEMMs in bf16 via
// v_wmma_f32_16x16x32_bf16 (wave32).  A tiles staged to LDS with
// GLOBAL_LOAD_ASYNC_TO_LDS_B128 (ASYNCcnt path, CDNA5 ISA).
// ---------------------------------------------------------------------------

typedef __attribute__((ext_vector_type(16))) __bf16 v16bf;
typedef __attribute__((ext_vector_type(8)))  float  v8f;
typedef int async_v4i __attribute__((vector_size(16)));   // builtin's pointee type

#define HWD   1024          // h*w
#define CD    512           // channels
#define BD    32            // batch
#define CPG   16            // channels per group (32 groups)
#define NPIX  (BD * HWD)    // 32768 global pixels

__device__ __forceinline__ unsigned short f2bf(float f) {
  unsigned int u = __float_as_uint(f);
  u += 0x7FFFu + ((u >> 16) & 1u);          // round-to-nearest-even
  return (unsigned short)(u >> 16);
}

// ---------------------------------------------------------------------------
// Async copy of one 16B chunk: global -> LDS (ASYNCcnt tracked).
// Builtin signature (from probe diagnostic): (v4i AS1*, v4i AS3*, imm, imm).
// ---------------------------------------------------------------------------
__device__ __forceinline__ void async_copy_b128(const unsigned short* g,
                                                unsigned short* l) {
#if defined(__has_builtin) && __has_builtin(__builtin_amdgcn_global_load_async_to_lds_b128)
  __builtin_amdgcn_global_load_async_to_lds_b128(
      (__attribute__((address_space(1))) async_v4i*)(void*)g,
      (__attribute__((address_space(3))) async_v4i*)(void*)l, 0, 0);
#else
  unsigned ldsoff = (unsigned)(unsigned long long)l;   // LDS byte address
  asm volatile("global_load_async_to_lds_b128 %0, %1, off"
               :: "v"(ldsoff), "v"(g) : "memory");
#endif
}

__device__ __forceinline__ void wait_async0() {
#if defined(__has_builtin) && __has_builtin(__builtin_amdgcn_s_wait_asynccnt)
  __builtin_amdgcn_s_wait_asynccnt(0);
#else
  asm volatile("s_wait_asynccnt 0x0" ::: "memory");
#endif
}

// Stage a 16-row x KW-col bf16 A tile into LDS (all 256 threads cooperate).
template <int KW>
__device__ __forceinline__ void stage_A(const unsigned short* __restrict__ g,
                                        int row0, int stride,
                                        unsigned short* l) {
  constexpr int C8 = KW / 8;                 // 16B chunks per row
  for (int c = threadIdx.x; c < 16 * C8; c += 256) {
    int row = c / C8;
    int col = (c % C8) * 8;
    async_copy_b128(g + (size_t)(row0 + row) * stride + col, l + row * KW + col);
  }
  wait_async0();
  __syncthreads();
}

union FragU { uint4 q[2]; v16bf bf; };

// Load one 16x32 bf16 WMMA operand fragment from a row-major u16 matrix
// (global or LDS).  ISA layout (CDNA5 ISA): lanes 0-15 hold row M=lane,
// K {0..7, 16..23}; lanes 16-31 hold row M=lane-16, K {8..15, 24..31}.
// Used for A (rows = M) and for B when B is stored N-major.
__device__ __forceinline__ v16bf load_frag_bf16(const unsigned short* __restrict__ base,
                                                int row0, int stride, int k0) {
  int lane = threadIdx.x & 31;
  int r    = lane & 15;
  int koff = (lane >> 4) << 3;              // 0 or 8
  const unsigned short* p = base + (size_t)(row0 + r) * stride + (k0 + koff);
  FragU u;
  u.q[0] = *(const uint4*)(p);              // K chunk +0..7
  u.q[1] = *(const uint4*)(p + 16);         // K chunk +16..23
  return u.bf;
}

// ---------------------------------------------------------------------------
// 1) GroupNorm statistics: one block per (batch, group), reduce 1024*16 elems.
// ---------------------------------------------------------------------------
__global__ void __launch_bounds__(256) k_gn_stats(const float* __restrict__ x,
                                                  float* __restrict__ mean,
                                                  float* __restrict__ rstd) {
  int bg = blockIdx.x;                       // b*32 + g
  int b = bg >> 5, g = bg & 31;
  const float* xb = x + (size_t)b * HWD * CD + g * CPG;
  float s = 0.f, ss = 0.f;
  for (int i = threadIdx.x; i < HWD * CPG; i += 256) {
    int p = i >> 4, ch = i & 15;
    float v = xb[(size_t)p * CD + ch];
    s += v; ss += v * v;
  }
  __shared__ float rs[256], rss[256];
  rs[threadIdx.x] = s; rss[threadIdx.x] = ss;
  __syncthreads();
  for (int off = 128; off > 0; off >>= 1) {
    if ((int)threadIdx.x < off) {
      rs[threadIdx.x]  += rs[threadIdx.x + off];
      rss[threadIdx.x] += rss[threadIdx.x + off];
    }
    __syncthreads();
  }
  if (threadIdx.x == 0) {
    const float inv = 1.0f / (HWD * CPG);
    float m   = rs[0] * inv;
    float var = rss[0] * inv - m * m;
    mean[bg] = m;
    rstd[bg] = rsqrtf(var + 1e-6f);
  }
}

// ---------------------------------------------------------------------------
// 2) Apply GroupNorm, cast to bf16.  float4-vectorized; a float4 never
//    crosses a 16-channel group boundary so the (b,g) index is uniform.
// ---------------------------------------------------------------------------
__global__ void __launch_bounds__(256) k_gn_apply(const float4* __restrict__ x4,
                                                  const float4* __restrict__ gamma4,
                                                  const float4* __restrict__ beta4,
                                                  const float* __restrict__ mean,
                                                  const float* __restrict__ rstd,
                                                  ushort4* __restrict__ xn4) {
  size_t total4 = (size_t)NPIX * CD / 4;
  for (size_t e = (size_t)blockIdx.x * 256 + threadIdx.x; e < total4;
       e += (size_t)gridDim.x * 256) {
    int c4   = (int)(e & (CD / 4 - 1));      // float4 index within channel dim
    int c    = c4 * 4;
    size_t pix = e / (CD / 4);
    int bg   = (int)(pix >> 10) * 32 + (c >> 4);
    float m = mean[bg], r = rstd[bg];
    float4 xv = x4[e], gv = gamma4[c4], bv = beta4[c4];
    ushort4 o;
    o.x = f2bf((xv.x - m) * r * gv.x + bv.x);
    o.y = f2bf((xv.y - m) * r * gv.y + bv.y);
    o.z = f2bf((xv.z - m) * r * gv.z + bv.z);
    o.w = f2bf((xv.w - m) * r * gv.w + bv.w);
    xn4[e] = o;
  }
}

// ---------------------------------------------------------------------------
// 3) Transpose + cast the four 512x512 weights to N-major bf16 (B operands).
// ---------------------------------------------------------------------------
__global__ void __launch_bounds__(256) k_cast_wT(const float* w0, const float* w1,
                                                 const float* w2, const float* w3,
                                                 unsigned short* t0, unsigned short* t1,
                                                 unsigned short* t2, unsigned short* t3) {
  const float* w; unsigned short* t;
  switch (blockIdx.y) {
    case 0:  w = w0; t = t0; break;
    case 1:  w = w1; t = t1; break;
    case 2:  w = w2; t = t2; break;
    default: w = w3; t = t3; break;
  }
  int e = blockIdx.x * 256 + threadIdx.x;    // 512*512 total
  int d = e >> 9, c = e & 511;
  t[e] = f2bf(w[c * CD + d]);                // wT[d*512 + c] = w[c,d]
}

// ---------------------------------------------------------------------------
// 4) QKV projection GEMM: (32768 x 512) * (512 x 512) per weight.
//    A tile async-staged in LDS; each wave: 16(M) x 64(N), 64 WMMAs.
//    Epilogue stores Q,K pre-transposed for the S GEMM; V stored flat.
// ---------------------------------------------------------------------------
__global__ void __launch_bounds__(256) k_qkv(const unsigned short* __restrict__ xn,
                                             const unsigned short* __restrict__ wqT,
                                             const unsigned short* __restrict__ wkT,
                                             const unsigned short* __restrict__ wvT,
                                             const float* __restrict__ bq,
                                             const float* __restrict__ bk,
                                             const float* __restrict__ bv,
                                             unsigned short* __restrict__ QT,
                                             unsigned short* __restrict__ KT,
                                             unsigned short* __restrict__ VB) {
  __shared__ unsigned short Atile[16 * CD];  // 16 KB
  int which = blockIdx.y;
  const unsigned short* wT = which == 0 ? wqT : (which == 1 ? wkT : wvT);
  const float* bias        = which == 0 ? bq  : (which == 1 ? bk  : bv);
  int m0   = blockIdx.x * 16;                // global pixel tile
  stage_A<CD>(xn, m0, CD, Atile);
  int wave = threadIdx.x >> 5;
  int n0   = wave * 64;
  const v8f zero = {0.f,0.f,0.f,0.f,0.f,0.f,0.f,0.f};
  v8f acc[4] = {zero, zero, zero, zero};
  for (int k0 = 0; k0 < CD; k0 += 32) {
    v16bf a = load_frag_bf16(Atile, 0, CD, k0);
#pragma unroll
    for (int t = 0; t < 4; ++t) {
      v16bf bfr = load_frag_bf16(wT, n0 + t * 16, CD, k0);
      acc[t] = __builtin_amdgcn_wmma_f32_16x16x32_bf16(false, a, false, bfr,
                                                       (short)0, acc[t], false, false);
    }
  }
  int lane  = threadIdx.x & 31;
  int ncol  = lane & 15;
  int mbase = (lane >> 4) << 3;
  for (int t = 0; t < 4; ++t) {
    int n = n0 + t * 16 + ncol;
    float bval = bias[n];
#pragma unroll
    for (int v = 0; v < 8; ++v) {
      int m = m0 + mbase + v;                // global pixel
      unsigned short h = f2bf(acc[t][v] + bval);
      if (which == 2) {
        VB[(size_t)m * CD + n] = h;          // flat == Vf row-major view
      } else {
        int bb = m >> 10, pp = m & 1023;
        int f  = pp * CD + n;                // within-batch flat index
        int i  = f >> 10, j = f & 1023;      // raw-reshape (512,1024) coords
        unsigned short* dst = (which == 0 ? QT : KT);
        dst[(size_t)bb * (CD * HWD) + (size_t)j * CD + i] = h;  // transposed
      }
    }
  }
}

// ---------------------------------------------------------------------------
// 5) S = scale * Qf^T Kf (per batch, 16-row tile), softmax fully in
//    registers: shfl_xor 16-lane row reductions + 1KB LDS cross-wave combine.
//    Each wave: 16 x 128 of S (8 WMMA tiles), 128 WMMAs.
// ---------------------------------------------------------------------------
__global__ void __launch_bounds__(256) k_attn_softmax(const unsigned short* __restrict__ QT,
                                                      const unsigned short* __restrict__ KT,
                                                      unsigned short* __restrict__ P) {
  __shared__ unsigned short Atile[16 * CD];  // 16 KB
  __shared__ float redmax[8 * 16], redsum[8 * 16];
  int b  = blockIdx.y;
  int p0 = blockIdx.x * 16;
  const unsigned short* qb = QT + (size_t)b * CD * HWD;
  const unsigned short* kb = KT + (size_t)b * CD * HWD;
  stage_A<CD>(qb, p0, CD, Atile);
  int wave = threadIdx.x >> 5;
  int n0   = wave * 128;
  const v8f zero = {0.f,0.f,0.f,0.f,0.f,0.f,0.f,0.f};
  v8f acc[8] = {zero, zero, zero, zero, zero, zero, zero, zero};
  for (int k0 = 0; k0 < CD; k0 += 32) {
    v16bf a = load_frag_bf16(Atile, 0, CD, k0);
#pragma unroll
    for (int t = 0; t < 8; ++t) {
      __builtin_prefetch(kb + (size_t)(n0 + t * 16) * CD + k0 + 32, 0, 1);
      v16bf bfr = load_frag_bf16(kb, n0 + t * 16, CD, k0);
      acc[t] = __builtin_amdgcn_wmma_f32_16x16x32_bf16(false, a, false, bfr,
                                                       (short)0, acc[t], false, false);
    }
  }
  const float scale = 0.044194173824159216f; // 512^-0.5
  int lane  = threadIdx.x & 31;
  int ncol  = lane & 15;
  int mbase = (lane >> 4) << 3;              // rows mbase..mbase+7
#pragma unroll
  for (int t = 0; t < 8; ++t)
#pragma unroll
    for (int v = 0; v < 8; ++v) acc[t][v] *= scale;

  // per-wave row max (16 lanes of one half-wave own one row's 128 columns)
#pragma unroll
  for (int v = 0; v < 8; ++v) {
    float m = acc[0][v];
#pragma unroll
    for (int t = 1; t < 8; ++t) m = fmaxf(m, acc[t][v]);
    for (int msk = 1; msk < 16; msk <<= 1) m = fmaxf(m, __shfl_xor(m, msk, 32));
    if (ncol == 0) redmax[wave * 16 + mbase + v] = m;
  }
  __syncthreads();
#pragma unroll
  for (int v = 0; v < 8; ++v) {
    float m = redmax[0 * 16 + mbase + v];
    for (int w = 1; w < 8; ++w) m = fmaxf(m, redmax[w * 16 + mbase + v]);
    float s = 0.f;
#pragma unroll
    for (int t = 0; t < 8; ++t) {
      float e = __expf(acc[t][v] - m);
      acc[t][v] = e;
      s += e;
    }
    for (int msk = 1; msk < 16; msk <<= 1) s += __shfl_xor(s, msk, 32);
    if (ncol == 0) redsum[wave * 16 + mbase + v] = s;
  }
  __syncthreads();
#pragma unroll
  for (int v = 0; v < 8; ++v) {
    float s = 0.f;
    for (int w = 0; w < 8; ++w) s += redsum[w * 16 + mbase + v];
    float inv = 1.0f / s;
    unsigned short* prow = P + (size_t)b * HWD * HWD + (size_t)(p0 + mbase + v) * HWD;
#pragma unroll
    for (int t = 0; t < 8; ++t)
      prow[n0 + t * 16 + ncol] = f2bf(acc[t][v] * inv);
  }
}

// ---------------------------------------------------------------------------
// 6) O = Vf @ P^T per batch: M=512, N=1024, K=1024.  P rows are directly the
//    N-major B operand.  O stored row-major (i*1024+p) == pixel-major view.
// ---------------------------------------------------------------------------
__global__ void __launch_bounds__(256) k_av(const unsigned short* __restrict__ VB,
                                            const unsigned short* __restrict__ P,
                                            unsigned short* __restrict__ OB) {
  __shared__ unsigned short Atile[16 * HWD]; // 32 KB
  int b    = blockIdx.z;
  int m0   = blockIdx.x * 16;                      // channel-row tile (i)
  const unsigned short* vb = VB + (size_t)b * CD * HWD;
  const unsigned short* pb = P  + (size_t)b * HWD * HWD;
  stage_A<HWD>(vb, m0, HWD, Atile);
  int wave = threadIdx.x >> 5;
  int n0   = blockIdx.y * 512 + wave * 64;         // attention position (p)
  const v8f zero = {0.f,0.f,0.f,0.f,0.f,0.f,0.f,0.f};
  v8f acc[4] = {zero, zero, zero, zero};
  for (int k0 = 0; k0 < HWD; k0 += 32) {
    v16bf a = load_frag_bf16(Atile, 0, HWD, k0);
#pragma unroll
    for (int t = 0; t < 4; ++t) {
      __builtin_prefetch(pb + (size_t)(n0 + t * 16) * HWD + k0 + 32, 0, 1);
      v16bf bfr = load_frag_bf16(pb, n0 + t * 16, HWD, k0);
      acc[t] = __builtin_amdgcn_wmma_f32_16x16x32_bf16(false, a, false, bfr,
                                                       (short)0, acc[t], false, false);
    }
  }
  int lane  = threadIdx.x & 31;
  int ncol  = lane & 15;
  int mbase = (lane >> 4) << 3;
  unsigned short* ob = OB + (size_t)b * CD * HWD;
  for (int t = 0; t < 4; ++t)
#pragma unroll
    for (int v = 0; v < 8; ++v)
      ob[(size_t)(m0 + mbase + v) * HWD + n0 + t * 16 + ncol] = f2bf(acc[t][v]);
}

// ---------------------------------------------------------------------------
// 7) Output projection + bias + residual, fp32 result.
//    OB reinterpreted pixel-major (raw reshape) is contiguous: A[m,k]=OB[m*512+k].
// ---------------------------------------------------------------------------
__global__ void __launch_bounds__(256) k_proj(const unsigned short* __restrict__ OB,
                                              const unsigned short* __restrict__ woT,
                                              const float* __restrict__ bo,
                                              const float* __restrict__ x,
                                              float* __restrict__ out) {
  __shared__ unsigned short Atile[16 * CD];  // 16 KB
  int m0   = blockIdx.x * 16;                // global pixel tile
  stage_A<CD>(OB, m0, CD, Atile);
  int wave = threadIdx.x >> 5;
  int n0   = wave * 64;
  const v8f zero = {0.f,0.f,0.f,0.f,0.f,0.f,0.f,0.f};
  v8f acc[4] = {zero, zero, zero, zero};
  for (int k0 = 0; k0 < CD; k0 += 32) {
    v16bf a = load_frag_bf16(Atile, 0, CD, k0);
#pragma unroll
    for (int t = 0; t < 4; ++t) {
      v16bf bfr = load_frag_bf16(woT, n0 + t * 16, CD, k0);
      acc[t] = __builtin_amdgcn_wmma_f32_16x16x32_bf16(false, a, false, bfr,
                                                       (short)0, acc[t], false, false);
    }
  }
  int lane  = threadIdx.x & 31;
  int ncol  = lane & 15;
  int mbase = (lane >> 4) << 3;
  for (int t = 0; t < 4; ++t)
#pragma unroll
    for (int v = 0; v < 8; ++v) {
      int n = n0 + t * 16 + ncol;
      size_t idx = (size_t)(m0 + mbase + v) * CD + n;
      out[idx] = acc[t][v] + bo[n] + x[idx];
    }
}

// ---------------------------------------------------------------------------
// Launch
// ---------------------------------------------------------------------------
extern "C" void kernel_launch(void* const* d_in, const int* in_sizes, int n_in,
                              void* d_out, int out_size, void* d_ws, size_t ws_size,
                              hipStream_t stream) {
  (void)in_sizes; (void)n_in; (void)out_size; (void)ws_size;
  const float* x     = (const float*)d_in[0];
  const float* gamma = (const float*)d_in[1];
  const float* beta  = (const float*)d_in[2];
  const float* wq    = (const float*)d_in[3];
  const float* bq    = (const float*)d_in[4];
  const float* wk    = (const float*)d_in[5];
  const float* bk    = (const float*)d_in[6];
  const float* wv    = (const float*)d_in[7];
  const float* bv    = (const float*)d_in[8];
  const float* wo    = (const float*)d_in[9];
  const float* bo    = (const float*)d_in[10];
  float* out = (float*)d_out;

  char* ws = (char*)d_ws;
  size_t off = 0;
  auto alloc = [&](size_t bytes) -> char* {
    char* p = ws + off;
    off += (bytes + 255) & ~(size_t)255;
    return p;
  };
  float*          mean = (float*)alloc(BD * 32 * sizeof(float));
  float*          rstd = (float*)alloc(BD * 32 * sizeof(float));
  unsigned short* xn   = (unsigned short*)alloc((size_t)NPIX * CD * 2);
  unsigned short* wqT  = (unsigned short*)alloc((size_t)CD * CD * 2);
  unsigned short* wkT  = (unsigned short*)alloc((size_t)CD * CD * 2);
  unsigned short* wvT  = (unsigned short*)alloc((size_t)CD * CD * 2);
  unsigned short* woT  = (unsigned short*)alloc((size_t)CD * CD * 2);
  unsigned short* QT   = (unsigned short*)alloc((size_t)BD * HWD * CD * 2);
  unsigned short* KT   = (unsigned short*)alloc((size_t)BD * HWD * CD * 2);
  unsigned short* VB   = (unsigned short*)alloc((size_t)BD * HWD * CD * 2);
  unsigned short* P    = (unsigned short*)alloc((size_t)BD * HWD * HWD * 2);
  unsigned short* OB   = (unsigned short*)alloc((size_t)BD * CD * HWD * 2);

  k_gn_stats   <<<dim3(BD * 32), 256, 0, stream>>>(x, mean, rstd);
  k_gn_apply   <<<dim3(8192),    256, 0, stream>>>((const float4*)x, (const float4*)gamma,
                                                   (const float4*)beta, mean, rstd,
                                                   (ushort4*)xn);
  k_cast_wT    <<<dim3(1024, 4), 256, 0, stream>>>(wq, wk, wv, wo, wqT, wkT, wvT, woT);
  k_qkv        <<<dim3(NPIX / 16, 3), 256, 0, stream>>>(xn, wqT, wkT, wvT, bq, bk, bv,
                                                        QT, KT, VB);
  k_attn_softmax<<<dim3(HWD / 16, BD), 256, 0, stream>>>(QT, KT, P);
  k_av         <<<dim3(CD / 16, 2, BD), 256, 0, stream>>>(VB, P, OB);
  k_proj       <<<dim3(NPIX / 16), 256, 0, stream>>>(OB, woT, bo, x, out);
}
